// MultiHeadAttention_70858370449466
// MI455X (gfx1250) — matrix operand order repaired and checked by reading, not compiled, not run on hardware
//
#include <hip/hip_runtime.h>
#include <hip/hip_bf16.h>
#include <math.h>

// ---------------------------------------------------------------------------
// MI455X (gfx1250) fused MHA block: QKV proj -> flash attention -> out proj
// -> residual -> LayerNorm.  All GEMM work goes through v_wmma_f32_16x16x32_bf16
// (bf16 inputs, f32 accumulation); softmax/LN statistics stay in f32.
// This revision software-pipelines all GEMM loops (double-buffered operands so
// s_wait_loadcnt before each WMMA targets loads issued a full step earlier)
// and prefetches the next KV block (global_prefetch_b8).
// Workspace: 4x 512KB bf16 transposed weights + 4x 16MB bf16 activations.
// ---------------------------------------------------------------------------

typedef __bf16 bf16_t;
typedef __attribute__((ext_vector_type(16))) __bf16 v16bf;
typedef __attribute__((ext_vector_type(8)))  __bf16 v8bf;
typedef __attribute__((ext_vector_type(8)))  float  v8f;

#define DEV __device__ __forceinline__

constexpr int Bd = 2, Ld = 8, Nd = 1024, Dd = 512, Hd = 8, HDd = 64;
constexpr int Md = Bd * Ld * Nd;          // 16384 token rows
constexpr float LN_EPS_F = 1e-5f;

DEV v8f vzero8() {
    v8f z;
#pragma unroll
    for (int i = 0; i < 8; ++i) z[i] = 0.0f;
    return z;
}

DEV v16bf combine8(v8bf a, v8bf b) {
    v16bf r;
#pragma unroll
    for (int i = 0; i < 8; ++i) { r[i] = a[i]; r[i + 8] = b[i]; }
    return r;
}

DEV v8f wmma_bf16(v16bf a, v16bf b, v8f c) {
    // D = A(16x32 bf16) * B(32x16 bf16) + C(16x16 f32)
    return __builtin_amdgcn_wmma_f32_16x16x32_bf16(
        /*neg_a=*/false, a, /*neg_b=*/false, b,
        /*c_mod=*/(short)0, c, /*reuse_a=*/false, /*reuse_b=*/false);
}

// A-operand (16x32 bf16) from f32 row: lane holds row M=lane&15,
// K = 16*hi + [0..15].
DEV v16bf load_a_f32(const float* p, int k0) {
    v16bf r;
#pragma unroll
    for (int i = 0; i < 8; ++i) r[i]     = (__bf16)p[k0 + i];
#pragma unroll
    for (int i = 0; i < 8; ++i) r[i + 8] = (__bf16)p[k0 + 8 + i];
    return r;
}

// A-operand from a bf16 row (two contiguous 16-byte chunks).
DEV v16bf load_a_bf16(const bf16_t* p, int k0) {
    return combine8(*(const v8bf*)(p + k0), *(const v8bf*)(p + k0 + 8));
}

// ---------------------------------------------------------------------------
// K0: weights f32 [in][out] -> bf16 transposed [out][in]
// ---------------------------------------------------------------------------
__global__ void wt_kernel(const float* Wq, const float* Wk, const float* Wv,
                          const float* Wo, bf16_t* WqT, bf16_t* WkT,
                          bf16_t* WvT, bf16_t* WoT) {
    int which = blockIdx.y;
    const float* W = (which == 0) ? Wq : (which == 1) ? Wk : (which == 2) ? Wv : Wo;
    bf16_t*      T = (which == 0) ? WqT : (which == 1) ? WkT : (which == 2) ? WvT : WoT;
    int idx = blockIdx.x * 256 + threadIdx.x;   // [0, D*D)
    int din = idx / Dd;                         // in-dim (row of W)
    int e   = idx % Dd;                         // out-dim (col of W)
    T[(size_t)e * Dd + din] = (bf16_t)W[idx];
}

// ---------------------------------------------------------------------------
// K1: QKV projections.  block = 128 threads (4 waves), each wave: 16 rows x 64
// cols, K loop 512 in steps of 32, double-buffered operands.  z selects q/k/v.
// V output is stored transposed [bl][d][n] so the attention PV B-operand is a
// contiguous 32-byte lane load.
// ---------------------------------------------------------------------------
__global__ void proj_kernel(const float* xq, const float* xk, const float* xv,
                            const bf16_t* WqT, const bf16_t* WkT, const bf16_t* WvT,
                            const float* bq, const float* bk, const float* bv,
                            bf16_t* qp, bf16_t* kp, bf16_t* vpT) {
    int z = blockIdx.z;
    const float*  X    = (z == 0) ? xq : (z == 1) ? xk : xv;
    const bf16_t* Wt   = (z == 0) ? WqT : (z == 1) ? WkT : WvT;
    const float*  bias = (z == 0) ? bq : (z == 1) ? bk : bv;

    int wave = threadIdx.x >> 5, lane = threadIdx.x & 31;
    int lo = lane & 15, hi = lane >> 4;
    int rowBase = blockIdx.x * 64 + wave * 16;
    int colBase = blockIdx.y * 64;

    const float*  xrow  = X + (size_t)(rowBase + lo) * Dd;
    const bf16_t* wbase = Wt + (size_t)(colBase + lo) * Dd + 16 * hi;

    v8f acc[4];
#pragma unroll
    for (int t = 0; t < 4; ++t) acc[t] = vzero8();

    // prologue: operands for k-step 0
    v16bf aCur = load_a_f32(xrow, 16 * hi);
    v16bf bCur[4];
#pragma unroll
    for (int t = 0; t < 4; ++t)
        bCur[t] = *(const v16bf*)(wbase + (size_t)16 * t * Dd);

    for (int ks = 0; ks < Dd / 32 - 1; ++ks) {
        // fetch k-step ks+1 while WMMAs of step ks are in flight
        v16bf aNxt = load_a_f32(xrow, (ks + 1) * 32 + 16 * hi);
        v16bf bNxt[4];
#pragma unroll
        for (int t = 0; t < 4; ++t)
            bNxt[t] = *(const v16bf*)(wbase + (size_t)16 * t * Dd + (ks + 1) * 32);
#pragma unroll
        for (int t = 0; t < 4; ++t) acc[t] = wmma_bf16(aCur, bCur[t], acc[t]);
        aCur = aNxt;
#pragma unroll
        for (int t = 0; t < 4; ++t) bCur[t] = bNxt[t];
    }
#pragma unroll
    for (int t = 0; t < 4; ++t) acc[t] = wmma_bf16(aCur, bCur[t], acc[t]);

    bf16_t* outN = (z == 0) ? qp : kp;
#pragma unroll
    for (int t = 0; t < 4; ++t) {
        int ocol = colBase + 16 * t + lo;
        float bb = bias[ocol];
#pragma unroll
        for (int r = 0; r < 8; ++r) {
            int m = rowBase + r + 8 * hi;       // global token row
            float o = acc[t][r] + bb;
            if (z == 2) {                       // V: [bl][d][n]
                int bl = m >> 10, n = m & (Nd - 1);
                vpT[((size_t)bl * Dd + ocol) * Nd + n] = (bf16_t)o;
            } else {                            // Q/K: [m][d]
                outN[(size_t)m * Dd + ocol] = (bf16_t)o;
            }
        }
    }
}

// ---------------------------------------------------------------------------
// K2: flash attention.  grid = (B*L*H, N/64); block = 128 threads (4 waves),
// each wave independently owns 16 q-rows.  Per 64-key block: V B-operands are
// hoisted to the loop top (resolve during QK^T + softmax), K tiles are
// double-buffered, next block is prefetched (global_prefetch_b8).  P is
// re-laid-out C->A through a per-wave LDS strip.
// ---------------------------------------------------------------------------
__global__ void attn_kernel(const bf16_t* qp, const bf16_t* kp, const bf16_t* vpT,
                            const unsigned char* mask, bf16_t* attnOut) {
    __shared__ bf16_t ldsP[4][16 * 64];         // 8 KB: per-wave P strip
    int wave = threadIdx.x >> 5, lane = threadIdx.x & 31;
    int lo = lane & 15, hi = lane >> 4;

    int combo = blockIdx.x;                     // (b*L + l)*H + h
    int h  = combo % Hd;
    int bl = combo / Hd;
    int b  = bl / Ld;
    int hd0 = h * HDd;
    int qRow0 = blockIdx.y * 64 + wave * 16;

    const bf16_t* qpBL = qp + (size_t)bl * Nd * Dd;
    const bf16_t* kpBL = kp + (size_t)bl * Nd * Dd;
    const bf16_t* vpBL = vpT + (size_t)bl * Dd * Nd;
    const unsigned char* mrow = mask + (size_t)b * Nd * Nd;
    const float scale = 0.125f;                 // HD^-0.5 = 1/8

    // Q A-operands: 2 k-steps of 32 covering HD=64, held for the whole loop.
    const bf16_t* qrow = qpBL + (size_t)(qRow0 + lo) * Dd + hd0;
    v16bf aQ[2];
#pragma unroll
    for (int s = 0; s < 2; ++s) aQ[s] = load_a_bf16(qrow, 32 * s + 16 * hi);

    float rowM[8], rowL[8];
    v8f acc[4];
#pragma unroll
    for (int r = 0; r < 8; ++r) { rowM[r] = -1e30f; rowL[r] = 0.0f; }
#pragma unroll
    for (int t = 0; t < 4; ++t) acc[t] = vzero8();

    bf16_t* pw = &ldsP[wave][0];

    for (int kb = 0; kb < Nd / 64; ++kb) {
        int kv0 = kb * 64;

        // ---- V B-operands for this block: independent of softmax, issue
        //      them first so they resolve behind QK^T + softmax ----
        v16bf bV[2][4];
#pragma unroll
        for (int s = 0; s < 2; ++s) {
#pragma unroll
            for (int t = 0; t < 4; ++t) {
                const bf16_t* vr =
                    vpBL + (size_t)(hd0 + 16 * t + lo) * Nd + kv0 + 32 * s + 16 * hi;
                bV[s][t] = *(const v16bf*)vr;
            }
        }

        // ---- prefetch next block's K/V rows into cache (global_prefetch) ----
        if (kb + 1 < Nd / 64) {
            int nv0 = kv0 + 64;
            __builtin_prefetch(kpBL + (size_t)(nv0 + lane) * Dd + hd0, 0, 1);
            __builtin_prefetch(kpBL + (size_t)(nv0 + 32 + lane) * Dd + hd0, 0, 1);
            __builtin_prefetch(vpBL + (size_t)(hd0 + 2 * lane) * Nd + nv0, 0, 1);
        }

        // ---- S = Q K^T for a 16x64 strip; K tiles double-buffered over t ----
        v8f st[4];
        const bf16_t* kr0 = kpBL + (size_t)(kv0 + lo) * Dd + hd0 + 16 * hi;
        v16bf kCur0 = *(const v16bf*)(kr0);
        v16bf kCur1 = *(const v16bf*)(kr0 + 32);
#pragma unroll
        for (int t = 0; t < 4; ++t) {
            v16bf kN0, kN1;
            if (t < 3) {
                const bf16_t* krn =
                    kpBL + (size_t)(kv0 + 16 * (t + 1) + lo) * Dd + hd0 + 16 * hi;
                kN0 = *(const v16bf*)(krn);
                kN1 = *(const v16bf*)(krn + 32);
            }
            v8f c = vzero8();
            c = wmma_bf16(aQ[0], kCur0, c);
            c = wmma_bf16(aQ[1], kCur1, c);
            st[t] = c;
            if (t < 3) { kCur0 = kN0; kCur1 = kN1; }
        }

        // ---- scale + mask (C layout: col = lane&15 + 16t, row = r + 8*hi) ----
#pragma unroll
        for (int t = 0; t < 4; ++t) {
            int key = kv0 + 16 * t + lo;
#pragma unroll
            for (int r = 0; r < 8; ++r) {
                float sv = st[t][r] * scale;
                int qr = qRow0 + r + 8 * hi;
                if (!mrow[(size_t)qr * Nd + key]) sv = -1e30f;
                st[t][r] = sv;
            }
        }

        // ---- online softmax, row stats per (r, half) ----
#pragma unroll
        for (int r = 0; r < 8; ++r) {
            float mb = fmaxf(fmaxf(st[0][r], st[1][r]), fmaxf(st[2][r], st[3][r]));
#pragma unroll
            for (int d = 8; d >= 1; d >>= 1) mb = fmaxf(mb, __shfl_xor(mb, d, 32));
            float mt = fmaxf(rowM[r], mb);
            float corr = __expf(rowM[r] - mt);
            float ps = 0.0f;
#pragma unroll
            for (int t = 0; t < 4; ++t) {
                float p = __expf(st[t][r] - mt);
                st[t][r] = p;
                ps += p;
            }
#pragma unroll
            for (int d = 8; d >= 1; d >>= 1) ps += __shfl_xor(ps, d, 32);
            rowL[r] = rowL[r] * corr + ps;
            rowM[r] = mt;
#pragma unroll
            for (int t = 0; t < 4; ++t) acc[t][r] *= corr;
        }

        // ---- P: C layout -> LDS row-major strip [16][64] (bf16) ----
#pragma unroll
        for (int t = 0; t < 4; ++t) {
#pragma unroll
            for (int r = 0; r < 8; ++r)
                pw[(r + 8 * hi) * 64 + 16 * t + lo] = (bf16_t)st[t][r];
        }

        // ---- O += P V  (A from LDS strip; V operands already resident) ----
        const bf16_t* prow = pw + lo * 64;
#pragma unroll
        for (int s = 0; s < 2; ++s) {
            v16bf aP = load_a_bf16(prow, 32 * s + 16 * hi);
#pragma unroll
            for (int t = 0; t < 4; ++t)
                acc[t] = wmma_bf16(aP, bV[s][t], acc[t]);
        }
    }

    // ---- normalize and store bf16 [m][d] for the output projection ----
#pragma unroll
    for (int r = 0; r < 8; ++r) {
        float inv = (rowL[r] > 0.0f) ? 1.0f / rowL[r] : 0.0f;
#pragma unroll
        for (int t = 0; t < 4; ++t) {
            float o = acc[t][r] * inv;
            attnOut[((size_t)bl * Nd + qRow0 + r + 8 * hi) * Dd + hd0 + 16 * t + lo] =
                (bf16_t)o;
        }
    }
}

// ---------------------------------------------------------------------------
// K3: out = LN(resid + attnOut @ Wo + bo).  block = 256 threads (8 waves),
// 16 rows x full 512 cols per block so LN sees whole rows; GEMM loop is
// double-buffered; result staged in LDS; per-row mean/var via wave32 shfl_xor.
// ---------------------------------------------------------------------------
__global__ void out_kernel(const bf16_t* attnOut, const bf16_t* WoT, const float* bo,
                           const float* resid, const float* gamma, const float* beta,
                           float* out) {
    __shared__ float xs[16 * Dd];               // 32 KB
    int wave = threadIdx.x >> 5, lane = threadIdx.x & 31;
    int lo = lane & 15, hi = lane >> 4;
    int rowBase = blockIdx.x * 16;
    int colBase = wave * 64;

    const bf16_t* arow  = attnOut + (size_t)(rowBase + lo) * Dd;
    const bf16_t* wbase = WoT + (size_t)(colBase + lo) * Dd + 16 * hi;

    v8f acc[4];
#pragma unroll
    for (int t = 0; t < 4; ++t) acc[t] = vzero8();

    v16bf aCur = load_a_bf16(arow, 16 * hi);
    v16bf bCur[4];
#pragma unroll
    for (int t = 0; t < 4; ++t)
        bCur[t] = *(const v16bf*)(wbase + (size_t)16 * t * Dd);

    for (int ks = 0; ks < Dd / 32 - 1; ++ks) {
        v16bf aNxt = load_a_bf16(arow, (ks + 1) * 32 + 16 * hi);
        v16bf bNxt[4];
#pragma unroll
        for (int t = 0; t < 4; ++t)
            bNxt[t] = *(const v16bf*)(wbase + (size_t)16 * t * Dd + (ks + 1) * 32);
#pragma unroll
        for (int t = 0; t < 4; ++t) acc[t] = wmma_bf16(aCur, bCur[t], acc[t]);
        aCur = aNxt;
#pragma unroll
        for (int t = 0; t < 4; ++t) bCur[t] = bNxt[t];
    }
#pragma unroll
    for (int t = 0; t < 4; ++t) acc[t] = wmma_bf16(aCur, bCur[t], acc[t]);

#pragma unroll
    for (int t = 0; t < 4; ++t) {
        int ocol = colBase + 16 * t + lo;
        float bb = bo[ocol];
#pragma unroll
        for (int r = 0; r < 8; ++r)
            xs[(r + 8 * hi) * Dd + ocol] = acc[t][r] + bb;
    }
    __syncthreads();

    // LayerNorm: each wave handles 2 of the 16 rows.
#pragma unroll
    for (int rr = 0; rr < 2; ++rr) {
        int lr = wave * 2 + rr;
        int grow = rowBase + lr;
        const float* qres = resid + (size_t)grow * Dd;
        float xv[16];
        float sum = 0.0f, sq = 0.0f;
#pragma unroll
        for (int i = 0; i < 16; ++i) {
            int c = lane + 32 * i;
            float x = xs[lr * Dd + c] + qres[c];
            xv[i] = x;
            sum += x;
            sq += x * x;
        }
#pragma unroll
        for (int d = 16; d >= 1; d >>= 1) {
            sum += __shfl_xor(sum, d, 32);
            sq  += __shfl_xor(sq, d, 32);
        }
        float mu = sum * (1.0f / Dd);
        float var = sq * (1.0f / Dd) - mu * mu;
        float rstd = rsqrtf(var + LN_EPS_F);
        float* orow = out + (size_t)grow * Dd;
#pragma unroll
        for (int i = 0; i < 16; ++i) {
            int c = lane + 32 * i;
            orow[c] = (xv[i] - mu) * rstd * gamma[c] + beta[c];
        }
    }
}

// ---------------------------------------------------------------------------
extern "C" void kernel_launch(void* const* d_in, const int* in_sizes, int n_in,
                              void* d_out, int out_size, void* d_ws, size_t ws_size,
                              hipStream_t stream) {
    const float*         q     = (const float*)d_in[0];
    const float*         k     = (const float*)d_in[1];
    const float*         v     = (const float*)d_in[2];
    const unsigned char* mask  = (const unsigned char*)d_in[3];
    const float*         Wq    = (const float*)d_in[4];
    const float*         bq    = (const float*)d_in[5];
    const float*         Wk    = (const float*)d_in[6];
    const float*         bk    = (const float*)d_in[7];
    const float*         Wv    = (const float*)d_in[8];
    const float*         bv    = (const float*)d_in[9];
    const float*         Wo    = (const float*)d_in[10];
    const float*         bo    = (const float*)d_in[11];
    const float*         gamma = (const float*)d_in[12];
    const float*         beta  = (const float*)d_in[13];

    char* ws = (char*)d_ws;
    const size_t wBytes   = (size_t)Dd * Dd * sizeof(bf16_t);   // 512 KB each
    const size_t actBytes = (size_t)Md * Dd * sizeof(bf16_t);   // 16 MB each
    bf16_t* WqT  = (bf16_t*)(ws + 0 * wBytes);
    bf16_t* WkT  = (bf16_t*)(ws + 1 * wBytes);
    bf16_t* WvT  = (bf16_t*)(ws + 2 * wBytes);
    bf16_t* WoT  = (bf16_t*)(ws + 3 * wBytes);
    char* p = ws + 4 * wBytes;
    bf16_t* qp    = (bf16_t*)p; p += actBytes;
    bf16_t* kp    = (bf16_t*)p; p += actBytes;
    bf16_t* vpT   = (bf16_t*)p; p += actBytes;
    bf16_t* attnO = (bf16_t*)p; p += actBytes;
    (void)in_sizes; (void)n_in; (void)out_size; (void)ws_size;

    dim3 g0(Dd * Dd / 256, 4);
    wt_kernel<<<g0, 256, 0, stream>>>(Wq, Wk, Wv, Wo, WqT, WkT, WvT, WoT);

    dim3 g1(Md / 64, Dd / 64, 3);
    proj_kernel<<<g1, 128, 0, stream>>>(q, k, v, WqT, WkT, WvT, bq, bk, bv,
                                        qp, kp, vpT);

    dim3 g2(Bd * Ld * Hd, Nd / 64);
    attn_kernel<<<g2, 128, 0, stream>>>(qp, kp, vpT, mask, attnO);

    out_kernel<<<Md / 16, 256, 0, stream>>>(attnO, WoT, bo, q, gamma, beta,
                                            (float*)d_out);
}
